// PitchShiftAug_31971736551657
// MI455X (gfx1250) — matrix operand order repaired and checked by reading, not compiled, not run on hardware
//
#include <hip/hip_runtime.h>
#include <stdint.h>

// ---------------------------------------------------------------------------
// Pitch-shift augmentation via DFT-as-GEMM on CDNA5 WMMA (gfx1250, wave32).
// Workspace requirement: ~194.5 MB (see offsets below).
// ---------------------------------------------------------------------------

typedef __attribute__((ext_vector_type(16))) _Float16 v16h;
typedef __attribute__((ext_vector_type(8)))  float    v8f;

#define T_LEN    160000
#define NFFT     2048
#define HOP      512
#define NBINS    1025
#define NFRAMES  313        // 1 + T/HOP
#define SMAX     354        // ceil(313 / 2^(-2/12)) + 2  (compile-time constant)
#define NBATCH   32

#define MPAD_F   320        // 313 -> 20 tiles of 16
#define NPAD_F   2112       // 2*NBINS = 2050 -> 132 tiles of 16 (= 33 groups of 4)
#define KPAD_I   2080       // 2050 -> 65 k-steps of 32
#define MPAD_I   368        // 354 -> 23 tiles of 16
#define YSTR_LEN 180736     // (SMAX-1)*HOP

// workspace layout (bytes, 256-aligned)
#define OFF_RATE 0u
#define OFF_BFWD 256u
#define SZ_BFWD  ((size_t)NPAD_F * NFFT * 2)              // 8,650,752
#define OFF_BINV (OFF_BFWD + SZ_BFWD)
#define SZ_BINV  ((size_t)NFFT * KPAD_I * 2)              // 8,519,680
#define OFF_AFWD (OFF_BINV + SZ_BINV)
#define SZ_AFWD  ((size_t)NBATCH * MPAD_F * NFFT * 2)     // 41,943,040
#define OFF_DMAT (OFF_AFWD + SZ_AFWD)
#define SZ_DMAT  ((size_t)NBATCH * MPAD_F * NPAD_F * 4)   // 86,507,520
#define OFF_AINV (OFF_DMAT + SZ_DMAT)
#define SZ_AINV  ((size_t)NBATCH * MPAD_I * KPAD_I * 2)   // 48,988,160
#define OFF_FT   OFF_AFWD   /* reuses Afwd+Dmat (dead after vocoder): 96.5MB<128.5MB */
#define OFF_YSTR OFF_AINV   /* reuses Ainv (dead after inverse GEMM): 23.1MB<49MB    */

#define TWO_PI_F 6.28318530717958647692f
#define TWO_PI_D 6.283185307179586476925286766559

__device__ __forceinline__ float hannw(int n) {
  return 0.5f - 0.5f * cosf(TWO_PI_F * (float)n / (float)NFFT);
}

// ---------------------------------------------------------------------------
// WMMA fragment loads (wave32); A and B share a contiguous k order, not the ISA table order.
// A (16x32 f16, MxK): lane L holds row M=L&15; K = (L>=16?16:0) + e.
__device__ __forceinline__ v16h load_frag_a(const _Float16* __restrict__ A, int lda,
                                            int m0, int k0, int lane) {
  const _Float16* p = A + (size_t)(m0 + (lane & 15)) * lda + k0 + ((lane >> 4) << 4);
  v16h r;
#pragma unroll
  for (int e = 0; e < 16; ++e)
    r[e] = p[e];
  return r;
}

// B (32x16 f16, KxN): lane L holds column N=L&15; K = (L>=16?16:0) + e.
// Bt is stored N-major ([n][k], row length = ldk) so this is 32B contiguous.
__device__ __forceinline__ v16h load_frag_b(const _Float16* __restrict__ Bt, int ldk,
                                            int k0, int n0, int lane) {
  const _Float16* p = Bt + (size_t)(n0 + (lane & 15)) * ldk + k0 + ((lane >> 4) << 4);
  v16h r;
#pragma unroll
  for (int e = 0; e < 16; ++e) r[e] = p[e];
  return r;
}

// C/D (16x16 f32): VGPR v -> M = (lane>=16?8:0)+v, N = lane&15.
__device__ __forceinline__ void store_tile(float* __restrict__ C, int ldc,
                                           int m0, int n0, int lane, v8f acc) {
  float* p = C + (size_t)(m0 + ((lane >> 4) << 3)) * ldc + n0 + (lane & 15);
#pragma unroll
  for (int v = 0; v < 8; ++v) p[(size_t)v * ldc] = acc[v];
}

// ---------------------------------------------------------------------------
// Batched GEMM: C[b] = A[b] (MxK, f16) x B (KxN, f16, shared, stored N-major).
// One wave per (batch, m-tile, group of 4 n-tiles); A fragment reused 4x.
// Nt is guaranteed to be a multiple of 4 -> no partial groups, no divergence.
// Wave id forced into SGPRs via readfirstlane -> pure scalar control flow,
// EXEC stays all-1s for every v_wmma (ISA 7.12 requirement).
__global__ __launch_bounds__(256)
void k_gemm_f16w(const _Float16* __restrict__ A, const _Float16* __restrict__ Bt,
                 float* __restrict__ C, int Mt, int Ngrp, int Ksteps,
                 int lda, int ldbk, int ldc, size_t strideA, size_t strideC) {
  int wid  = __builtin_amdgcn_readfirstlane(
                 (int)((blockIdx.x * blockDim.x + threadIdx.x) >> 5));
  int lane = (int)(threadIdx.x & 31);
  int perB = Mt * Ngrp;
  if (wid >= NBATCH * perB) return;           // scalar branch; EXEC untouched
  int b  = wid / perB;
  int r  = wid - b * perB;
  int mt = r / Ngrp;
  int ng = r - mt * Ngrp;
  int m0 = mt << 4;
  int n0 = ng << 6;                           // 4 n-tiles of 16 per wave
  const _Float16* Ab = A + strideA * (size_t)b;
  float*          Cb = C + strideC * (size_t)b;
  v8f a0 = {}, a1 = {}, a2 = {}, a3 = {};
  for (int ks = 0; ks < Ksteps; ++ks) {
    int k0 = ks << 5;
    if (ks + 1 < Ksteps)  // emits global_prefetch_b8 on gfx1250
      __builtin_prefetch(Ab + (size_t)(m0 + (lane & 15)) * lda + k0 + 32, 0, 1);
    v16h af = load_frag_a(Ab, lda, m0, k0, lane);
    v16h b0 = load_frag_b(Bt, ldbk, k0, n0,      lane);
    v16h b1 = load_frag_b(Bt, ldbk, k0, n0 + 16, lane);
    v16h b2 = load_frag_b(Bt, ldbk, k0, n0 + 32, lane);
    v16h b3 = load_frag_b(Bt, ldbk, k0, n0 + 48, lane);
    a0 = __builtin_amdgcn_wmma_f32_16x16x32_f16(false, af, false, b0, (short)0, a0, false, false);
    a1 = __builtin_amdgcn_wmma_f32_16x16x32_f16(false, af, false, b1, (short)0, a1, false, false);
    a2 = __builtin_amdgcn_wmma_f32_16x16x32_f16(false, af, false, b2, (short)0, a2, false, false);
    a3 = __builtin_amdgcn_wmma_f32_16x16x32_f16(false, af, false, b3, (short)0, a3, false, false);
  }
  store_tile(Cb, ldc, m0, n0,      lane, a0);
  store_tile(Cb, ldc, m0, n0 + 16, lane, a1);
  store_tile(Cb, ldc, m0, n0 + 32, lane, a2);
  store_tile(Cb, ldc, m0, n0 + 48, lane, a3);
}

// ---------------------------------------------------------------------------
__global__ void k_rate(const float* __restrict__ log_lims, const float* __restrict__ u,
                       float* __restrict__ rate) {
  int b = (int)threadIdx.x;
  if (b >= NBATCH) return;
  float l0 = 2.f / (1.f + expf(-log_lims[0])) - 1.f;
  float l1 = 2.f / (1.f + expf(-log_lims[1])) - 1.f;
  float n_steps = 1.f + (u[b] * (l1 - l0) + l0);
  rate[b] = exp2f(-n_steps / 12.f);
}

// Forward DFT matrix, stored N-major: Bt[n][k].
// n<1025: cos(2*pi*n*k/N); 1025<=n<2050: -sin(2*pi*(n-1025)*k/N); else 0.
__global__ void k_fill_bfwd(_Float16* __restrict__ Bt) {
  int idx = (int)(blockIdx.x * blockDim.x + threadIdx.x);
  if (idx >= NPAD_F * NFFT) return;
  int n = idx / NFFT, k = idx - n * NFFT;
  float v = 0.f;
  if (n < NBINS) {
    int ph = (n * k) & (NFFT - 1);                  // exact angle reduction
    v = cosf((float)ph * (TWO_PI_F / NFFT));
  } else if (n < 2 * NBINS) {
    int ph = ((n - NBINS) * k) & (NFFT - 1);
    v = -sinf((float)ph * (TWO_PI_F / NFFT));
  }
  Bt[idx] = (_Float16)v;
}

// Inverse DFT matrix (irfft basis), stored N-major: Bt[n][k2] = B[k2][n].
__global__ void k_fill_binv(_Float16* __restrict__ Bt) {
  int idx = (int)(blockIdx.x * blockDim.x + threadIdx.x);
  if (idx >= NFFT * KPAD_I) return;
  int n = idx / KPAD_I, r = idx - n * KPAD_I;
  float v = 0.f;
  if (r < NBINS) {
    int ph = (n * r) & (NFFT - 1);
    v = cosf((float)ph * (TWO_PI_F / NFFT));
  } else if (r < 2 * NBINS) {
    int ph = (n * (r - NBINS)) & (NFFT - 1);
    v = -sinf((float)ph * (TWO_PI_F / NFFT));
  }
  Bt[idx] = (_Float16)v;
}

// Windowed frames with inline reflect padding; rows >= NFRAMES are zero.
__global__ void k_frames(const float* __restrict__ x, _Float16* __restrict__ Af) {
  int idx = (int)(blockIdx.x * blockDim.x + threadIdx.x);
  if (idx >= NBATCH * MPAD_F * NFFT) return;
  int n  = idx & (NFFT - 1);
  int fm = (idx / NFFT) % MPAD_F;
  int b  = idx / (NFFT * MPAD_F);
  float v = 0.f;
  if (fm < NFRAMES) {
    int j = fm * HOP + n - (NFFT / 2);
    if (j < 0) j = -j;                               // reflect (no edge repeat)
    if (j >= T_LEN) j = 2 * T_LEN - 2 - j;
    v = x[(size_t)b * T_LEN + j] * hannw(n);
  }
  Af[idx] = (_Float16)v;
}

__global__ void k_zero16(uint4* __restrict__ p, long long n16) {
  long long i = (long long)blockIdx.x * blockDim.x + threadIdx.x;
  if (i < n16) p[i] = make_uint4(0u, 0u, 0u, 0u);
}

// Phase vocoder: one thread per (batch, bin). Sequential cumsum over SMAX
// frames; phase kept wrapped in double. Writes irfft-weighted Re/Im into Ainv.
__global__ void k_vocoder(const float* __restrict__ D, _Float16* __restrict__ S,
                          const float* __restrict__ rateArr) {
  int tid = (int)(blockIdx.x * blockDim.x + threadIdx.x);
  if (tid >= NBATCH * NBINS) return;
  int b = tid / NBINS, k = tid - b * NBINS;
  float rate = rateArr[b];
  const float* Db = D + (size_t)b * MPAD_F * NPAD_F;
  _Float16*    Sb = S + (size_t)b * MPAD_I * KPAD_I;
  double phi_adv = (double)k * (TWO_PI_D / 4.0);     // k*pi/2 (linspace step)
  double phase = atan2((double)Db[NBINS + k], (double)Db[k]);  // angle(D[:,0])
  float wk = (k == 0 || k == NBINS - 1) ? (1.0f / NFFT) : (2.0f / NFFT);
  for (int s = 0; s < SMAX; ++s) {
    float t  = (float)s * rate;
    float ft = floorf(t);
    int idx = (int)ft; if (idx > NFRAMES - 1) idx = NFRAMES - 1;
    float alpha = t - ft;
    const float* r0 = Db + (size_t)idx * NPAD_F;
    const float* r1 = Db + (size_t)(idx + 1) * NPAD_F;   // row NFRAMES == 0 (Dpad)
    float c0r = r0[k], c0i = r0[NBINS + k];
    float c1r = r1[k], c1i = r1[NBINS + k];
    float mag = (1.f - alpha) * sqrtf(c0r * c0r + c0i * c0i)
              +         alpha * sqrtf(c1r * c1r + c1i * c1i);
    float ph = (float)phase;
    Sb[(size_t)s * KPAD_I + k]         = (_Float16)(wk * mag * cosf(ph));
    Sb[(size_t)s * KPAD_I + NBINS + k] = (_Float16)(wk * mag * sinf(ph));
    double dph = atan2((double)c1i, (double)c1r)
               - atan2((double)c0i, (double)c0r) - phi_adv;
    dph  -= TWO_PI_D * rint(dph / TWO_PI_D);         // wrap to (-pi, pi]
    phase += phi_adv + dph;                          // step = phi_adv + dphase
    phase -= TWO_PI_D * rint(phase / TWO_PI_D);      // keep wrapped (mod 2pi exact)
  }
}

// Overlap-add + win^2 normalize + center slice, gather form (<=4 frames/sample).
__global__ void k_ystr(const float* __restrict__ Ft, float* __restrict__ Ystr) {
  int idx = (int)(blockIdx.x * blockDim.x + threadIdx.x);
  if (idx >= NBATCH * YSTR_LEN) return;
  int b = idx / YSTR_LEN, q = idx - b * YSTR_LEN;
  int p = q + NFFT / 2;
  int f_hi = p >> 9; if (f_hi > SMAX - 1) f_hi = SMAX - 1;
  int f_lo = (p - (NFFT - 1) + (HOP - 1)) >> 9; if (f_lo < 0) f_lo = 0;
  const float* Fb = Ft + (size_t)b * MPAD_I * NFFT;
  float num = 0.f, den = 0.f;
  for (int f = f_lo; f <= f_hi; ++f) {
    int n = p - (f << 9);
    float w = hannw(n);
    num += Fb[(size_t)f * NFFT + n] * w;
    den += w * w;
  }
  Ystr[(size_t)b * YSTR_LEN + q] = num / fmaxf(den, 1e-8f);
}

// Linear resample back to original length.
__global__ void k_resample(const float* __restrict__ Ystr, const float* __restrict__ rateArr,
                           float* __restrict__ out) {
  int idx = (int)(blockIdx.x * blockDim.x + threadIdx.x);
  if (idx >= NBATCH * T_LEN) return;
  int b = idx / T_LEN, i = idx - b * T_LEN;
  float pos = (float)i / rateArr[b];
  int i0 = (int)floorf(pos);
  if (i0 < 0) i0 = 0;
  if (i0 > YSTR_LEN - 2) i0 = YSTR_LEN - 2;
  float frac = pos - (float)i0;
  const float* Yb = Ystr + (size_t)b * YSTR_LEN;
  out[idx] = Yb[i0] * (1.f - frac) + Yb[i0 + 1] * frac;
}

// ---------------------------------------------------------------------------
extern "C" void kernel_launch(void* const* d_in, const int* in_sizes, int n_in,
                              void* d_out, int out_size, void* d_ws, size_t ws_size,
                              hipStream_t stream) {
  const float* x        = (const float*)d_in[0];
  const float* log_lims = (const float*)d_in[1];
  const float* u        = (const float*)d_in[2];
  float* out = (float*)d_out;
  char*  ws  = (char*)d_ws;

  float*     rate = (float*)(ws + OFF_RATE);
  _Float16*  Bfwd = (_Float16*)(ws + OFF_BFWD);
  _Float16*  Binv = (_Float16*)(ws + OFF_BINV);
  _Float16*  Afwd = (_Float16*)(ws + OFF_AFWD);
  float*     Dmat = (float*)(ws + OFF_DMAT);
  _Float16*  Ainv = (_Float16*)(ws + OFF_AINV);
  float*     Ft   = (float*)(ws + OFF_FT);
  float*     Ystr = (float*)(ws + OFF_YSTR);

  k_rate<<<1, 32, 0, stream>>>(log_lims, u, rate);
  k_fill_bfwd<<<(NPAD_F * NFFT + 255) / 256, 256, 0, stream>>>(Bfwd);
  k_fill_binv<<<(NFFT * KPAD_I + 255) / 256, 256, 0, stream>>>(Binv);
  k_frames<<<(NBATCH * MPAD_F * NFFT + 255) / 256, 256, 0, stream>>>(x, Afwd);

  {   // forward DFT GEMM: [32] 320x2048 @ 2048x2112 -> Dmat
    int Mt = MPAD_F / 16, Ngrp = (NPAD_F / 16) / 4, Ks = NFFT / 32;
    int waves = NBATCH * Mt * Ngrp;
    k_gemm_f16w<<<(waves * 32 + 255) / 256, 256, 0, stream>>>(
        Afwd, Bfwd, Dmat, Mt, Ngrp, Ks,
        NFFT, NFFT, NPAD_F, (size_t)MPAD_F * NFFT, (size_t)MPAD_F * NPAD_F);
  }

  k_zero16<<<(int)((SZ_AINV / 16 + 255) / 256), 256, 0, stream>>>(
      (uint4*)Ainv, (long long)(SZ_AINV / 16));
  k_vocoder<<<(NBATCH * NBINS + 255) / 256, 256, 0, stream>>>(Dmat, Ainv, rate);

  {   // inverse DFT GEMM: [32] 368x2080 @ 2080x2048 -> Ft (overlays Afwd/Dmat)
    int Mt = MPAD_I / 16, Ngrp = (NFFT / 16) / 4, Ks = KPAD_I / 32;
    int waves = NBATCH * Mt * Ngrp;
    k_gemm_f16w<<<(waves * 32 + 255) / 256, 256, 0, stream>>>(
        Ainv, Binv, Ft, Mt, Ngrp, Ks,
        KPAD_I, KPAD_I, NFFT, (size_t)MPAD_I * KPAD_I, (size_t)MPAD_I * NFFT);
  }

  k_ystr<<<(NBATCH * YSTR_LEN + 255) / 256, 256, 0, stream>>>(Ft, Ystr);
  k_resample<<<(NBATCH * T_LEN + 255) / 256, 256, 0, stream>>>(Ystr, rate, out);
}
